// GRUModel_79963701117389
// MI455X (gfx1250) — matrix operand order repaired and checked by reading, not compiled, not run on hardware
//
#include <hip/hip_runtime.h>
#include <hip/hip_bf16.h>

// ---------------------------------------------------------------------------
// Problem constants (match reference)
// ---------------------------------------------------------------------------
#define BSZ   32
#define TLEN  1000
#define HSZ   512
#define NMEL  128
#define OSZ   61
#define BT    (BSZ * TLEN)          // 32000 rows, divisible by 16

// cluster scan geometry: 1 cluster per direction, 8 WGs x 4 waves
#define CLU_WGS   8                 // workgroups per direction-cluster
#define SC_WAVES  4                 // waves per workgroup (1 per SIMD of a WGP)

#if __has_attribute(amdgpu_cluster_dims)
#define CLUSTER_DIMS __attribute__((amdgpu_cluster_dims(CLU_WGS, 1, 1)))
#else
#define CLUSTER_DIMS
#endif

typedef __bf16 bf16_t;
typedef __attribute__((ext_vector_type(16))) __bf16 v16bf;
typedef __attribute__((ext_vector_type(8)))  __bf16 v8bf;
typedef __attribute__((ext_vector_type(8)))  float  v8f;

// ---------------------------------------------------------------------------
// WMMA helper: D = A(16x32 bf16) * B(32x16 bf16) + C (f32)
// ---------------------------------------------------------------------------
__device__ __forceinline__ v8f wmma_bf16(v16bf a, v16bf b, v8f c) {
  return __builtin_amdgcn_wmma_f32_16x16x32_bf16(
      /*neg_a=*/false, a, /*neg_b=*/false, b,
      /*c_mod=*/(short)0, c, /*reuse_a=*/false, /*reuse_b=*/false);
}

// A-matrix 16x32 bf16 layout (ISA 7.12.2):
//   lanes 0-15  : M = lane,    K = 0..7  (v0..3) and 16..23 (v4..7)
//   lanes 16-31 : M = lane-16, K = 8..15 (v0..3) and 24..31 (v4..7)
// => per lane: two contiguous 8-element chunks at K = half*8 and 16+half*8
__device__ __forceinline__ v16bf load_bf16_A(const bf16_t* base, int ld, int lane) {
  const int m    = lane & 15;
  const int half = lane >> 4;
  const bf16_t* p = base + (size_t)m * ld + half * 8;
  v8bf lo = *(const v8bf*)(p);
  v8bf hi = *(const v8bf*)(p + 16);
  v16bf a;
#pragma unroll
  for (int e = 0; e < 8; ++e) { a[e] = lo[e]; a[e + 8] = hi[e]; }
  return a;
}

// B-matrix 32x16 bf16 layout: lane n = lane&15 holds K = half*8..+7 and
// 16+half*8..+7 (same K order as A).  Since B = W^T, lane n's
// K-runs are row-slices of the row-major weight matrix W[n][k...].
__device__ __forceinline__ v16bf load_bf16_B(const bf16_t* base, int ld, int lane) {
  const int n    = lane & 15;
  const int half = lane >> 4;
  const bf16_t* p = base + (size_t)n * ld + half * 8;
  v8bf lo = *(const v8bf*)(p);
  v8bf hi = *(const v8bf*)(p + 16);
  v16bf b;
#pragma unroll
  for (int e = 0; e < 8; ++e) { b[e] = lo[e]; b[e + 8] = hi[e]; }
  return b;
}

__device__ __forceinline__ float sigmoidf_(float x) {
  return 1.0f / (1.0f + __expf(-x));
}

// ---------------------------------------------------------------------------
// fp32 -> bf16 conversion helpers
// ---------------------------------------------------------------------------
__global__ void cvt_bf16_kernel(const float* __restrict__ in,
                                bf16_t* __restrict__ out, long n) {
  long i = (long)blockIdx.x * blockDim.x + threadIdx.x;
  if (i < n) out[i] = (bf16_t)in[i];
}

// fc_w [61,1024] -> bf16 padded [64,1024] (rows 61..63 zero)
__global__ void cvt_fc_kernel(const float* __restrict__ in,
                              bf16_t* __restrict__ out) {
  int i = blockIdx.x * blockDim.x + threadIdx.x;   // 0 .. 64*1024-1
  int r = i >> 10;
  int c = i & 1023;
  out[i] = (r < OSZ) ? (bf16_t)in[r * 1024 + c] : (bf16_t)0.0f;
}

// ---------------------------------------------------------------------------
// Input-projection GEMM:  gx[dir][m][g] = x[m] . w_ih[dir][g] + b_ih[dir][g]
//   x   : [BT, I] bf16          (row m = b*T + t, natural time order)
//   wih : [2, 3H, I] bf16
//   gx  : [2, BT, 3H] fp32
// one wave per 16x16 output tile (codegen verified: clause'd b128 + wmma)
// ---------------------------------------------------------------------------
__global__ __launch_bounds__(32)
void gx_gemm_kernel(const bf16_t* __restrict__ x,
                    const bf16_t* __restrict__ wih,
                    const float*  __restrict__ bih,
                    float*        __restrict__ gx,
                    int I) {
  const int mt   = blockIdx.x;       // 0..BT/16-1
  const int nt   = blockIdx.y;       // 0..(3H/16)-1
  const int dir  = blockIdx.z;       // 0..1
  const int lane = threadIdx.x;
  const int half = lane >> 4;

  v8f acc = {};
  const bf16_t* A  = x   + (size_t)mt * 16 * I;
  const bf16_t* Wb = wih + ((size_t)dir * 3 * HSZ + (size_t)nt * 16) * I;

  for (int k = 0; k < I; k += 32) {
    v16bf a = load_bf16_A(A + k, I, lane);
    v16bf b = load_bf16_B(Wb + k, I, lane);
    acc = wmma_bf16(a, b, acc);
  }

  const int n = nt * 16 + (lane & 15);
  const float bias = bih[(size_t)dir * 3 * HSZ + n];
  float* O = gx + ((size_t)dir * BT + (size_t)mt * 16) * (3 * HSZ);
#pragma unroll
  for (int i = 0; i < 8; ++i) {
    const int m = half * 8 + i;                       // C layout: M = i + 8*half
    O[(size_t)m * (3 * HSZ) + n] = acc[i] + bias;
  }
}

// ---------------------------------------------------------------------------
// Recurrent scan across a WORKGROUP CLUSTER.
//
// Grid = 2 clusters (one per direction) x CLU_WGS workgroups x SC_WAVES waves.
// Each WG owns 64 h-columns (wave -> 16 columns); with 4 waves/WG each SIMD
// of a WGP carries exactly one wave, so:
//   * per-SIMD matrix work per step is only 96 WMMA, and
//   * each wave can afford ~500 VGPRs -> the wave's entire W_hh slice
//     (3 gates x 16 k-steps x 32x16 bf16 tile = 384 VGPRs) is preloaded into
//     registers ONCE and re-used for all 1000 timesteps.  The per-step k-loop
//     then issues only 32 A-tile loads (L2-hot h state) feeding 96 WMMAs.
//
// h state (32x512 bf16, double-buffered per dir) lives in global/L2, shared
// cluster-wide.  Per step:
//   prefetch gx(t+1) -> compute gh from hg[p] -> gates -> store to hg[p^1]
//   __threadfence() (release) -> s_cluster_barrier -> __threadfence() (acquire)
//
//   gx  : [2, BT, 3H] fp32  (natural time; reverse dir indexes flipped t)
//   whh : [2, 3H, H] bf16
//   out : [B, T, 2H] bf16   (dir d writes columns [d*H, d*H+H))
//   hg  : [2 dirs][2 bufs][B*H] bf16
// ---------------------------------------------------------------------------
CLUSTER_DIMS
__attribute__((amdgpu_waves_per_eu(1)))
__global__ __launch_bounds__(SC_WAVES * 32)
void gru_scan_cluster_kernel(const float*  __restrict__ gx,
                             const bf16_t* __restrict__ whh,
                             const float*  __restrict__ bhh,
                             bf16_t*       __restrict__ out,
                             bf16_t*       __restrict__ hg) {
  const int blk  = blockIdx.x;            // 0 .. 2*CLU_WGS-1
  const int dir  = blk / CLU_WGS;
  const int wg   = blk % CLU_WGS;         // workgroup-in-cluster
  const int tid  = threadIdx.x;
  const int wid  = tid >> 5;              // 0..SC_WAVES-1
  const int lane = tid & 31;
  const int half = lane >> 4;
  const int nb   = (wg * SC_WAVES + wid) * 16;   // this wave's h column base
  const int nl   = nb + (lane & 15);             // this lane's h column

  bf16_t* h0 = hg + (size_t)dir * 2 * BSZ * HSZ;   // per-dir double buffer

  // ---- preload this wave's recurrent weights into registers (384 VGPRs) ----
  const bf16_t* W = whh + (size_t)dir * 3 * HSZ * HSZ;
  v16bf wr[16], wz[16], wn[16];
#pragma unroll
  for (int k = 0; k < 16; ++k) {
    wr[k] = load_bf16_B(W + ((size_t)(0 * HSZ) + nb) * HSZ + k * 32, HSZ, lane);
    wz[k] = load_bf16_B(W + ((size_t)(1 * HSZ) + nb) * HSZ + k * 32, HSZ, lane);
    wn[k] = load_bf16_B(W + ((size_t)(2 * HSZ) + nb) * HSZ + k * 32, HSZ, lane);
  }

  // cooperative zero of h(t=0) across the whole cluster
  for (int i = wg * (SC_WAVES * 32) + tid; i < BSZ * HSZ;
       i += CLU_WGS * SC_WAVES * 32) {
    h0[i] = (bf16_t)0.0f;
  }
  __threadfence();
  __builtin_amdgcn_s_cluster_barrier();
  __threadfence();

  const float* bh = bhh + (size_t)dir * 3 * HSZ;
  const float bhr = bh[nl];
  const float bhz = bh[HSZ + nl];
  const float bhn = bh[2 * HSZ + nl];
  const float* G = gx + (size_t)dir * BT * (3 * HSZ);

  int p = 0;
  for (int t = 0; t < TLEN; ++t) {
    const int te = dir ? (TLEN - 1 - t) : t;    // original-time index
    const bf16_t* hp_g = h0 + (size_t)p * BSZ * HSZ;
    bf16_t*       hn_g = h0 + (size_t)(p ^ 1) * BSZ * HSZ;

    // prefetch next step's input-gate lines (1 cacheline per gate; lane = row)
    if (t + 1 < TLEN) {
      const int tn = dir ? (TLEN - 2 - t) : (t + 1);
      const float* gn = G + ((size_t)lane * TLEN + tn) * (3 * HSZ);
      __builtin_prefetch(gn + nb, 0, 0);                 // global_prefetch_b8
      __builtin_prefetch(gn + HSZ + nb, 0, 0);
      __builtin_prefetch(gn + 2 * HSZ + nb, 0, 0);
    }

    v8f accr[2] = {}, accz[2] = {}, accn[2] = {};

#pragma unroll
    for (int k = 0; k < 16; ++k) {
      v16bf a0 = load_bf16_A(hp_g + k * 32,            HSZ, lane);  // batch 0-15
      v16bf a1 = load_bf16_A(hp_g + 16 * HSZ + k * 32, HSZ, lane);  // batch 16-31
      accr[0] = wmma_bf16(a0, wr[k], accr[0]);
      accr[1] = wmma_bf16(a1, wr[k], accr[1]);
      accz[0] = wmma_bf16(a0, wz[k], accz[0]);
      accz[1] = wmma_bf16(a1, wz[k], accz[1]);
      accn[0] = wmma_bf16(a0, wn[k], accn[0]);
      accn[1] = wmma_bf16(a1, wn[k], accn[1]);
    }

    // gate math in fp32; each lane owns column nl for 16 of the 32 batch rows
#pragma unroll
    for (int mt = 0; mt < 2; ++mt) {
#pragma unroll
      for (int i = 0; i < 8; ++i) {
        const int m = mt * 16 + half * 8 + i;          // batch row (C layout)
        const float* g = G + ((size_t)m * TLEN + te) * (3 * HSZ);
        const float hp = (float)hp_g[m * HSZ + nl];
        const float r  = sigmoidf_(g[nl]          + accr[mt][i] + bhr);
        const float z  = sigmoidf_(g[HSZ + nl]    + accz[mt][i] + bhz);
        const float nn = tanhf   (g[2 * HSZ + nl] + r * (accn[mt][i] + bhn));
        const float h  = (1.0f - z) * nn + z * hp;
        hn_g[m * HSZ + nl] = (bf16_t)h;
        out[((size_t)m * TLEN + te) * (2 * HSZ) + (size_t)dir * HSZ + nl] =
            (bf16_t)h;
      }
    }

    __threadfence();                         // release: flush h stores to L2
    __builtin_amdgcn_s_cluster_barrier();    // s_barrier_signal/-wait -3
    __threadfence();                         // acquire: invalidate stale L0
    p ^= 1;
  }
}

// ---------------------------------------------------------------------------
// Final FC + length mask:  out[b,t,o] = mask(t<len[b]) * (h2 . fc_w[o] + fc_b[o])
//   h2 : [BT, 1024] bf16 ; fw : [64, 1024] bf16 (rows 61..63 zero)
// block = 4 waves; wave w -> N tile w; blockIdx.x -> M tile
// ---------------------------------------------------------------------------
__global__ __launch_bounds__(128)
void fc_mask_kernel(const bf16_t* __restrict__ h2,
                    const bf16_t* __restrict__ fw,
                    const float*  __restrict__ fb,
                    const int*    __restrict__ lengths,
                    float*        __restrict__ out) {
  const int mt   = blockIdx.x;        // 0..BT/16-1
  const int wid  = threadIdx.x >> 5;  // 0..3
  const int lane = threadIdx.x & 31;
  const int half = lane >> 4;

  v8f acc = {};
  const bf16_t* A  = h2 + (size_t)mt * 16 * 1024;
  const bf16_t* Wb = fw + (size_t)wid * 16 * 1024;
  for (int k = 0; k < 1024; k += 32) {
    acc = wmma_bf16(load_bf16_A(A + k, 1024, lane),
                    load_bf16_B(Wb + k, 1024, lane), acc);
  }

  const int n = wid * 16 + (lane & 15);
  if (n < OSZ) {
    const float bias = fb[n];
#pragma unroll
    for (int i = 0; i < 8; ++i) {
      const int m = mt * 16 + half * 8 + i;   // row = b*T + t
      const int b = m / TLEN;
      const int t = m % TLEN;
      const float v = (t < lengths[b]) ? (acc[i] + bias) : 0.0f;
      out[(size_t)m * OSZ + n] = v;
    }
  }
}

// ---------------------------------------------------------------------------
// Host orchestration
// ---------------------------------------------------------------------------
static inline size_t align256(size_t x) { return (x + 255) & ~(size_t)255; }

extern "C" void kernel_launch(void* const* d_in, const int* in_sizes, int n_in,
                              void* d_out, int out_size, void* d_ws, size_t ws_size,
                              hipStream_t stream) {
  (void)in_sizes; (void)n_in; (void)out_size; (void)ws_size;

  const float* batch   = (const float*)d_in[0];   // [B,T,128]
  const int*   lengths = (const int*)  d_in[1];   // [B]
  const float* w_ih0   = (const float*)d_in[2];   // [2,3H,128]
  const float* w_hh0   = (const float*)d_in[3];   // [2,3H,H]
  const float* b_ih0   = (const float*)d_in[4];   // [2,3H]
  const float* b_hh0   = (const float*)d_in[5];   // [2,3H]
  const float* w_ih12  = (const float*)d_in[6];   // [2,2,3H,2H]
  const float* w_hh12  = (const float*)d_in[7];   // [2,2,3H,H]
  const float* b_ih12  = (const float*)d_in[8];   // [2,2,3H]
  const float* b_hh12  = (const float*)d_in[9];   // [2,2,3H]
  const float* fc_w    = (const float*)d_in[10];  // [61,2H]
  const float* fc_b    = (const float*)d_in[11];  // [61]
  float* out = (float*)d_out;                     // [B,T,61]

  // workspace carve-up (bf16 copies + fp32 gx + bf16 layer ping-pong)
  char* ws = (char*)d_ws;
  size_t off = 0;
  auto carve = [&](size_t bytes) { char* p = ws + off; off += align256(bytes); return p; };

  bf16_t* xbf     = (bf16_t*)carve((size_t)BT * NMEL * 2);                 // 8.2 MB
  bf16_t* wih0b   = (bf16_t*)carve((size_t)2 * 3 * HSZ * NMEL * 2);
  bf16_t* whh0b   = (bf16_t*)carve((size_t)2 * 3 * HSZ * HSZ * 2);
  bf16_t* wih12b  = (bf16_t*)carve((size_t)4 * 3 * HSZ * 2 * HSZ * 2);
  bf16_t* whh12b  = (bf16_t*)carve((size_t)4 * 3 * HSZ * HSZ * 2);
  bf16_t* fwb     = (bf16_t*)carve((size_t)64 * 1024 * 2);
  bf16_t* hstate  = (bf16_t*)carve((size_t)2 * 2 * BSZ * HSZ * 2);         // 128 KB
  float*  gxbuf   = (float*) carve((size_t)2 * BT * 3 * HSZ * 4);          // 393 MB
  bf16_t* hbufA   = (bf16_t*)carve((size_t)BT * 2 * HSZ * 2);              // 65.5 MB
  bf16_t* hbufB   = (bf16_t*)carve((size_t)BT * 2 * HSZ * 2);              // 65.5 MB

  auto cvt = [&](const float* src, bf16_t* dst, long n) {
    cvt_bf16_kernel<<<(unsigned)((n + 255) / 256), 256, 0, stream>>>(src, dst, n);
  };

  // --- convert weights / input to bf16 ---
  cvt(batch,  xbf,    (long)BT * NMEL);
  cvt(w_ih0,  wih0b,  (long)2 * 3 * HSZ * NMEL);
  cvt(w_hh0,  whh0b,  (long)2 * 3 * HSZ * HSZ);
  cvt(w_ih12, wih12b, (long)4 * 3 * HSZ * 2 * HSZ);
  cvt(w_hh12, whh12b, (long)4 * 3 * HSZ * HSZ);
  cvt_fc_kernel<<<(64 * 1024) / 256, 256, 0, stream>>>(fc_w, fwb);

  const dim3 gxGrid(BT / 16, (3 * HSZ) / 16, 2);
  const dim3 scanGrid(2 * CLU_WGS);
  const int  scanBlk = SC_WAVES * 32;

  // --- layer 0: input 128 -> hbufA ---
  gx_gemm_kernel<<<gxGrid, 32, 0, stream>>>(xbf, wih0b, b_ih0, gxbuf, NMEL);
  gru_scan_cluster_kernel<<<scanGrid, scanBlk, 0, stream>>>(
      gxbuf, whh0b, b_hh0, hbufA, hstate);

  // --- layer 1: input 1024 -> hbufB ---
  gx_gemm_kernel<<<gxGrid, 32, 0, stream>>>(hbufA, wih12b, b_ih12, gxbuf, 2 * HSZ);
  gru_scan_cluster_kernel<<<scanGrid, scanBlk, 0, stream>>>(
      gxbuf, whh12b, b_hh12, hbufB, hstate);

  // --- layer 2: input 1024 -> hbufA ---
  gx_gemm_kernel<<<gxGrid, 32, 0, stream>>>(hbufB,
                                            wih12b + (size_t)2 * 3 * HSZ * 2 * HSZ,
                                            b_ih12 + (size_t)2 * 3 * HSZ,
                                            gxbuf, 2 * HSZ);
  gru_scan_cluster_kernel<<<scanGrid, scanBlk, 0, stream>>>(
      gxbuf,
      whh12b + (size_t)2 * 3 * HSZ * HSZ,
      b_hh12 + (size_t)2 * 3 * HSZ,
      hbufA, hstate);

  // --- final FC + length mask ---
  fc_mask_kernel<<<BT / 16, 128, 0, stream>>>(hbufA, fwb, fc_b, lengths, out);
}
